// FlaxLlamaAttention_18305150616341
// MI455X (gfx1250) — matrix operand order repaired and checked by reading, not compiled, not run on hardware
//
#include <hip/hip_runtime.h>

typedef _Float16 f16;
typedef __attribute__((ext_vector_type(16))) _Float16 v16h;
typedef __attribute__((ext_vector_type(8)))  _Float16 v8h;
typedef __attribute__((ext_vector_type(8)))  float    v8f;

#define NEG_INF -3.0e38f

// ---------------------------------------------------------------------------
// f32 -> f16 elementwise convert
// ---------------------------------------------------------------------------
__global__ void cvt_f32_f16(const float* __restrict__ in, f16* __restrict__ out, int n) {
  int i = blockIdx.x * blockDim.x + threadIdx.x;
  if (i < n) out[i] = (f16)in[i];
}

// ---------------------------------------------------------------------------
// Transpose-convert: in f32 [K][N] row-major -> out f16 [N][K] row-major.
// 32x32 tile via LDS for coalesced reads and writes.
// ---------------------------------------------------------------------------
__global__ __launch_bounds__(256) void transpose_f32_to_f16(
    const float* __restrict__ in, f16* __restrict__ out, int K, int N) {
  __shared__ float tile[32][33];
  const int nb = blockIdx.x * 32, kb = blockIdx.y * 32;
  const int tx = threadIdx.x, ty = threadIdx.y;  // (32, 8)
#pragma unroll
  for (int i = 0; i < 32; i += 8)
    tile[ty + i][tx] = in[(size_t)(kb + ty + i) * N + nb + tx];
  __syncthreads();
#pragma unroll
  for (int i = 0; i < 32; i += 8)
    out[(size_t)(nb + ty + i) * K + kb + tx] = (f16)tile[tx][ty + i];
}

// ---------------------------------------------------------------------------
// WMMA GEMM: C(f32, MxN) = A(f16, MxK) * Bt(f16, [N][K]) with f32 accumulate.
// 256 threads = 8 waves; workgroup tile 128(M) x 64(N); each wave 32x32
// (2x2 grid of v_wmma_f32_16x16x32_f16). All dims multiples of tile sizes.
// Fragment layouts used here (A/B in contiguous k order, same for both):
//   A 16x32: lane L -> row M = L%16; halves 0..7 = K(16*(L/16)+i),
//            halves 8..15 = K(16*(L/16)+8+i)
//   B 32x16: lane L -> col N = L%16; halves j = K(16*(L/16)+j)
//   C 16x16: VGPR r, lane L -> (M = r+8*(L/16), N = L%16)
// ---------------------------------------------------------------------------
__global__ __launch_bounds__(256) void gemm_f16(
    const f16* __restrict__ A, const f16* __restrict__ Bt, float* __restrict__ C,
    int M, int N, int K) {
  const int lane = threadIdx.x & 31;
  const int wave = threadIdx.x >> 5;
  const int mw = wave & 3, nw = wave >> 2;              // 4 (M) x 2 (N) waves
  const int mbase = blockIdx.y * 128 + mw * 32;
  const int nbase = blockIdx.x * 64 + nw * 32;
  const int half = lane >> 4;
  const int l16 = lane & 15;

  const f16* arow0 = A + (size_t)(mbase + l16) * K;
  const f16* arow1 = A + (size_t)(mbase + 16 + l16) * K;
  const f16* brow0 = Bt + (size_t)(nbase + l16) * K;
  const f16* brow1 = Bt + (size_t)(nbase + 16 + l16) * K;

  v8f acc00 = {}, acc01 = {}, acc10 = {}, acc11 = {};

  for (int kb = 0; kb < K; kb += 32) {
    v8h lo0 = *(const v8h*)(arow0 + kb + 16 * half);
    v8h hi0 = *(const v8h*)(arow0 + kb + 8 + 16 * half);
    v8h lo1 = *(const v8h*)(arow1 + kb + 16 * half);
    v8h hi1 = *(const v8h*)(arow1 + kb + 8 + 16 * half);
    v16h bb0 = *(const v16h*)(brow0 + kb + 16 * half);
    v16h bb1 = *(const v16h*)(brow1 + kb + 16 * half);
    v16h a0, a1;
#pragma unroll
    for (int i = 0; i < 8; ++i) {
      a0[i] = lo0[i]; a0[8 + i] = hi0[i];
      a1[i] = lo1[i]; a1[8 + i] = hi1[i];
    }
    acc00 = __builtin_amdgcn_wmma_f32_16x16x32_f16(false, a0, false, bb0, (short)0, acc00, false, false);
    acc01 = __builtin_amdgcn_wmma_f32_16x16x32_f16(false, a0, false, bb1, (short)0, acc01, false, false);
    acc10 = __builtin_amdgcn_wmma_f32_16x16x32_f16(false, a1, false, bb0, (short)0, acc10, false, false);
    acc11 = __builtin_amdgcn_wmma_f32_16x16x32_f16(false, a1, false, bb1, (short)0, acc11, false, false);
  }

#pragma unroll
  for (int r = 0; r < 8; ++r) {
    const int m0 = mbase + r + 8 * half;
    C[(size_t)m0 * N + nbase + l16]             = acc00[r];
    C[(size_t)m0 * N + nbase + 16 + l16]        = acc01[r];
    C[(size_t)(m0 + 16) * N + nbase + l16]      = acc10[r];
    C[(size_t)(m0 + 16) * N + nbase + 16 + l16] = acc11[r];
  }
}

// ---------------------------------------------------------------------------
// RoPE + split/convert: QKV f32 [S][1536] -> Qh f16 [S][1024],
// Kh f16 [S][256] (roped), Vh f16 [S][256]. One block per sequence position.
// ---------------------------------------------------------------------------
__global__ __launch_bounds__(128) void rope_kernel(
    const float* __restrict__ qkv, const int* __restrict__ pos_ids,
    f16* __restrict__ Qh, f16* __restrict__ Kh, f16* __restrict__ Vh) {
  const int s = blockIdx.x;
  const int t = threadIdx.x;
  const float pos = (float)pos_ids[s];
  const float* base = qkv + (size_t)s * 1536;

  for (int idx = t; idx < 1024; idx += 128) {
    const int d = idx & 63, i = d & 31;
    const float freq = pos * __powf(10000.0f, -(float)i * (1.0f / 32.0f));
    float sn, cs;
    __sincosf(freq, &sn, &cs);
    const float x = base[idx];
    const float partner = (d < 32) ? -base[idx + 32] : base[idx - 32];
    Qh[(size_t)s * 1024 + idx] = (f16)(x * cs + partner * sn);
  }
  for (int idx = t; idx < 256; idx += 128) {
    const int d = idx & 63, i = d & 31;
    const float freq = pos * __powf(10000.0f, -(float)i * (1.0f / 32.0f));
    float sn, cs;
    __sincosf(freq, &sn, &cs);
    const float x = base[1024 + idx];
    const float partner = (d < 32) ? -base[1024 + idx + 32] : base[1024 + idx - 32];
    Kh[(size_t)s * 256 + idx] = (f16)(x * cs + partner * sn);
    Vh[(size_t)s * 256 + idx] = (f16)base[1280 + idx];
  }
}

// ---------------------------------------------------------------------------
// Sliding-window (W=64) GQA attention, one wave32 per (16-query tile, head).
// Key slab: 5 aligned 16-key tiles covering kpos in [s0-64, s0+15].
// QK^T and PV use v_wmma_f32_16x16x32_f16; softmax via __shfl_xor over the
// 16-lane half-groups; P re-laid-out via LDS into A-fragment form; V staged
// zero-padded in LDS so padded keys contribute 0 while staying in the softmax
// denominator (matches the reference's zero-padding semantics).
// ---------------------------------------------------------------------------
__global__ __launch_bounds__(32) void attn_kernel(
    const f16* __restrict__ Qh, const f16* __restrict__ Kh,
    const f16* __restrict__ Vh, const int* __restrict__ amask,
    f16* __restrict__ Oh) {
  __shared__ f16 ldsV[96 * 64];   // keys (slab rows, zero-padded) x dims
  __shared__ f16 ldsP[16 * 96];   // probs: queries x key-slab (zero-padded)

  const int qt = blockIdx.x;          // query tile (0..127)
  const int h  = blockIdx.y;          // head (0..15)
  const int hk = h >> 2;              // kv head
  const int s0 = qt * 16;
  const int lane = threadIdx.x;
  const int half = lane >> 4, l16 = lane & 15;
  const int kbase0 = s0 - 64;         // first key position of slab

  // zero LDS (padding rows/cols must be 0)
  {
    uint32_t* p = (uint32_t*)ldsV;
    for (int i = lane; i < 96 * 64 / 2; i += 32) p[i] = 0u;
    uint32_t* q = (uint32_t*)ldsP;
    for (int i = lane; i < 16 * 96 / 2; i += 32) q[i] = 0u;
  }
  __syncthreads();

  // stage valid V rows (16B chunks, coalesced)
  for (int c = lane; c < 640; c += 32) {
    const int row = c >> 3, off = (c & 7) * 8;
    const int kpos = kbase0 + row;
    if (kpos >= 0)
      *(v8h*)(ldsV + row * 64 + off) =
          *(const v8h*)(Vh + (size_t)kpos * 256 + hk * 64 + off);
  }
  __syncthreads();

  // Q fragments (two 32-wide D-chunks), A-layout, contiguous 16B loads
  const f16* qrow = Qh + (size_t)(s0 + l16) * 1024 + h * 64;
  v16h qa[2];
#pragma unroll
  for (int c = 0; c < 2; ++c) {
    v8h lo = *(const v8h*)(qrow + 32 * c + 16 * half);
    v8h hi = *(const v8h*)(qrow + 32 * c + 8 + 16 * half);
#pragma unroll
    for (int i = 0; i < 8; ++i) { qa[c][i] = lo[i]; qa[c][8 + i] = hi[i]; }
  }

  // scores for 5 key tiles
  float sc[5][8];
#pragma unroll
  for (int t = 0; t < 5; ++t) {
    const int kt = qt - 4 + t;
    v8f c = {};
    if (kt >= 0) {
      const f16* krow = Kh + (size_t)(kt * 16 + l16) * 256 + hk * 64;
#pragma unroll
      for (int ch = 0; ch < 2; ++ch) {
        v16h kb = *(const v16h*)(krow + 32 * ch + 16 * half);
        c = __builtin_amdgcn_wmma_f32_16x16x32_f16(false, qa[ch], false, kb, (short)0, c, false, false);
      }
    }
#pragma unroll
    for (int r = 0; r < 8; ++r) {
      const int m = r + 8 * half;          // query row in tile
      const int s = s0 + m;
      const int kpos = kt * 16 + l16;
      const int w = kpos - s + 63;
      float v;
      if (w < 0 || w > 63) {
        v = NEG_INF;                       // outside sliding window
      } else {
        const float dot = (kpos >= 0) ? c[r] * 0.125f : 0.0f;  // padded key -> 0
        v = (amask[s * 64 + w] > 0) ? dot : NEG_INF;
      }
      sc[t][r] = v;
    }
  }

  // row softmax over the 80-score slab (each row lives in one 16-lane group)
  float pm[5][8], wsum[8];
#pragma unroll
  for (int r = 0; r < 8; ++r) {
    float mx = NEG_INF;
#pragma unroll
    for (int t = 0; t < 5; ++t) mx = fmaxf(mx, sc[t][r]);
#pragma unroll
    for (int off = 1; off < 16; off <<= 1) mx = fmaxf(mx, __shfl_xor(mx, off, 16));
    float sum = 0.f;
#pragma unroll
    for (int t = 0; t < 5; ++t) { float p = __expf(sc[t][r] - mx); pm[t][r] = p; sum += p; }
#pragma unroll
    for (int off = 1; off < 16; off <<= 1) sum += __shfl_xor(sum, off, 16);
    wsum[r] = 1.0f / sum;
  }

  // P (C-layout) -> LDS (row-major 16 x 96, cols 80..95 stay zero)
#pragma unroll
  for (int t = 0; t < 5; ++t)
#pragma unroll
    for (int r = 0; r < 8; ++r)
      ldsP[(r + 8 * half) * 96 + t * 16 + l16] = (f16)(pm[t][r] * wsum[r]);
  __syncthreads();

  // PV: 3 K-chunks of 32 keys x 4 N-tiles of 16 dims
  v8f oacc[4] = {{}, {}, {}, {}};
#pragma unroll
  for (int c = 0; c < 3; ++c) {
    v16h pa;
    const f16* prow = ldsP + l16 * 96 + 32 * c + 16 * half;
#pragma unroll
    for (int i = 0; i < 8; ++i) { pa[i] = prow[i]; pa[8 + i] = prow[8 + i]; }
#pragma unroll
    for (int nt = 0; nt < 4; ++nt) {
      v16h vb;
      const f16* vcol = ldsV + (32 * c + 16 * half) * 64 + nt * 16 + l16;
#pragma unroll
      for (int j = 0; j < 16; ++j) vb[j] = vcol[j * 64];
      oacc[nt] = __builtin_amdgcn_wmma_f32_16x16x32_f16(false, pa, false, vb, (short)0, oacc[nt], false, false);
    }
  }

  // store attention output tile as f16 for the O-projection GEMM
#pragma unroll
  for (int nt = 0; nt < 4; ++nt)
#pragma unroll
    for (int r = 0; r < 8; ++r) {
      const int m = r + 8 * half;
      Oh[(size_t)(s0 + m) * 1024 + h * 64 + nt * 16 + l16] = (f16)oacc[nt][r];
    }
}

// ---------------------------------------------------------------------------
// launch
// ---------------------------------------------------------------------------
extern "C" void kernel_launch(void* const* d_in, const int* in_sizes, int n_in,
                              void* d_out, int out_size, void* d_ws, size_t ws_size,
                              hipStream_t stream) {
  const float* hs    = (const float*)d_in[0];   // (1,2048,1024) f32
  const int*   amask = (const int*)d_in[1];     // (1,2048,1,64) i32
  const int*   pos   = (const int*)d_in[2];     // (1,2048) i32
  const float* wq    = (const float*)d_in[3];   // (1024,1024)
  const float* wk    = (const float*)d_in[4];   // (1024,256)
  const float* wv    = (const float*)d_in[5];   // (1024,256)
  const float* wo    = (const float*)d_in[6];   // (1024,1024)
  float* out = (float*)d_out;                   // (1,2048,1024) f32

  char* ws = (char*)d_ws;
  f16* Xh     = (f16*)ws;  ws += (size_t)2048 * 1024 * 2;
  f16* WqkvT  = (f16*)ws;  ws += (size_t)1536 * 1024 * 2;   // [N=1536][K=1024]
  f16* WoT    = (f16*)ws;  ws += (size_t)1024 * 1024 * 2;   // [N=1024][K=1024]
  float* QKV  = (float*)ws; ws += (size_t)2048 * 1536 * 4;
  f16* Qh     = (f16*)ws;  ws += (size_t)2048 * 1024 * 2;
  f16* Kh     = (f16*)ws;  ws += (size_t)2048 * 256 * 2;
  f16* Vh     = (f16*)ws;  ws += (size_t)2048 * 256 * 2;
  f16* Ah     = (f16*)ws;  ws += (size_t)2048 * 1024 * 2;

  cvt_f32_f16<<<(2048 * 1024) / 256, 256, 0, stream>>>(hs, Xh, 2048 * 1024);

  dim3 tb(32, 8);
  transpose_f32_to_f16<<<dim3(1024 / 32, 1024 / 32), tb, 0, stream>>>(wq, WqkvT, 1024, 1024);
  transpose_f32_to_f16<<<dim3(256 / 32, 1024 / 32), tb, 0, stream>>>(wk, WqkvT + (size_t)1024 * 1024, 1024, 256);
  transpose_f32_to_f16<<<dim3(256 / 32, 1024 / 32), tb, 0, stream>>>(wv, WqkvT + (size_t)1280 * 1024, 1024, 256);
  transpose_f32_to_f16<<<dim3(1024 / 32, 1024 / 32), tb, 0, stream>>>(wo, WoT, 1024, 1024);

  // fused QKV projection: (2048 x 1536) = Xh (2048x1024) * WqkvT^T
  gemm_f16<<<dim3(1536 / 64, 2048 / 128), 256, 0, stream>>>(Xh, WqkvT, QKV, 2048, 1536, 1024);

  rope_kernel<<<2048, 128, 0, stream>>>(QKV, pos, Qh, Kh, Vh);

  attn_kernel<<<dim3(128, 16), 32, 0, stream>>>(Qh, Kh, Vh, amask, Ah);

  // output projection straight into d_out
  gemm_f16<<<dim3(1024 / 64, 2048 / 128), 256, 0, stream>>>(Ah, WoT, out, 2048, 1024, 1024);
}
